// GraphAttentionLayer_16260746183319
// MI455X (gfx1250) — matrix operand order repaired and checked by reading, not compiled, not run on hardware
//
#include <hip/hip_runtime.h>

// ---------------------------------------------------------------------------
// GAT layer for MI455X (gfx1250, wave32).
//   B=32, N=1024, F_IN=F_OUT=64, leaky-relu slope 0.2, elu output.
// Fused flash-style softmax: logits e_ij = leakyrelu(f1_i + f2_j) are computed
// on the fly (never materialized), shifted by the upper bound
// m_i = leakyrelu(f1_i + max_j f2_j)  (valid shift, exp<=1 fits f16), and
// P @ h is accumulated with v_wmma_f32_16x16x32_f16.
// Adjacency (33.5 MB) is streamed exactly once; h chunks are double-buffered
// in LDS via GLOBAL_LOAD_ASYNC_TO_LDS_B128 (ASYNCcnt).
// f1/f2 pre-scaled by log2(e) (leakyrelu is positively homogeneous) so the
// softmax needs one v_exp_f32 per element; leakyrelu is a single v_max
// (max(s, 0.2s)); the adjacency mask is a v_cvt_f32_ubyte + v_mul.
// ---------------------------------------------------------------------------

typedef __attribute__((ext_vector_type(16))) _Float16 v16h;
typedef __attribute__((ext_vector_type(8)))  _Float16 v8h;
typedef __attribute__((ext_vector_type(8)))  float    v8f;
typedef __attribute__((ext_vector_type(4)))  float    v4f;
typedef int v4i_vs __attribute__((vector_size(16)));   // matches builtin param

#define BB 32
#define NN 1024
#define FF 64
#define SLOPE 0.2f
#define LOG2E 1.4426950408889634f

// ---- global -> LDS async copy (16 B), double-buffer friendly --------------
__device__ __forceinline__ void async_b128(const void* gsrc, void* ldst) {
#if __has_builtin(__builtin_amdgcn_global_load_async_to_lds_b128)
  __builtin_amdgcn_global_load_async_to_lds_b128(
      (__attribute__((address_space(1))) v4i_vs*)(v4i_vs*)gsrc,
      (__attribute__((address_space(3))) v4i_vs*)(v4i_vs*)ldst, 0, 0);
#else
  *(v8h*)ldst = *(const v8h*)gsrc;            // sync fallback (VGPR roundtrip)
#endif
}

__device__ __forceinline__ void wait_async_lds() {
#if __has_builtin(__builtin_amdgcn_s_wait_asynccnt)
  __builtin_amdgcn_s_wait_asynccnt(0);
#elif __has_builtin(__builtin_amdgcn_global_load_async_to_lds_b128)
  asm volatile("s_wait_asynccnt 0" ::: "memory");
#endif
}

// ---------------------------------------------------------------------------
// Phase 1: h = X @ W (per-thread row GEMV, W staged in LDS).
// Writes hT[b][f][n] as f16 (transposed: contiguous-K layout for the WMMA B
// operand), and f1 = h.a1, f2 = h.a2 as f32.
// ---------------------------------------------------------------------------
__global__ __launch_bounds__(256) void gat_phase1(
    const float* __restrict__ X, const float* __restrict__ W,
    const float* __restrict__ a,
    _Float16* __restrict__ hT, float* __restrict__ f1, float* __restrict__ f2)
{
  __shared__ alignas(16) float Ws[FF * FF];
  __shared__ float a1s[FF], a2s[FF];
  const int t = threadIdx.x;
  for (int i = t; i < FF * FF; i += 256) Ws[i] = W[i];
  if (t < FF) { a1s[t] = a[t]; a2s[t] = a[FF + t]; }
  __syncthreads();

  const int row = blockIdx.x * 256 + t;       // 0 .. B*N-1
  const int b = row >> 10, n = row & (NN - 1);

  float x[FF];
  const float* xp = X + (size_t)row * FF;
#pragma unroll
  for (int k = 0; k < FF; ++k) x[k] = xp[k];

  float s1 = 0.f, s2 = 0.f;
  _Float16* hTp = hT + (size_t)b * FF * NN + n;
#pragma unroll 2
  for (int f0 = 0; f0 < FF; f0 += 4) {
    v4f hv = {0.f, 0.f, 0.f, 0.f};
#pragma unroll
    for (int k = 0; k < FF; ++k) {
      v4f w = *(const v4f*)&Ws[k * FF + f0];
      hv += x[k] * w;                          // LDS broadcast reads
    }
#pragma unroll
    for (int q = 0; q < 4; ++q) {
      hTp[(size_t)(f0 + q) * NN] = (_Float16)hv[q];
      s1 = fmaf(hv[q], a1s[f0 + q], s1);
      s2 = fmaf(hv[q], a2s[f0 + q], s2);
    }
  }
  f1[row] = s1;
  f2[row] = s2;
}

// ---------------------------------------------------------------------------
// Phase 1b: per-batch max of f2 (softmax shift bound).
// ---------------------------------------------------------------------------
__global__ __launch_bounds__(256) void gat_f2max(
    const float* __restrict__ f2, float* __restrict__ f2max)
{
  __shared__ float red[256];
  const int b = blockIdx.x, t = threadIdx.x;
  const float* p = f2 + b * NN;
  float m = p[t];
  for (int i = t + 256; i < NN; i += 256) m = fmaxf(m, p[i]);
  red[t] = m;
  __syncthreads();
  for (int s = 128; s > 0; s >>= 1) {
    if (t < s) red[t] = fmaxf(red[t], red[t + s]);
    __syncthreads();
  }
  if (t == 0) f2max[b] = red[0];
}

// ---------------------------------------------------------------------------
// Phase 2: fused masked-softmax attention + P @ h via WMMA.
// Block = 256 threads = 8 waves; wave w owns rows [i0, i0+16). grid=(B, N/128).
// Per 32-wide j-chunk the block stages the 64x32 f16 hT chunk into LDS
// (double-buffered, async), so h[b] is pulled from L2 once per block.
//
// WMMA operand layouts (CDNA5 ISA):
//   A 16x32 f16 : lane L -> row M=L%16, halves e=0..15 -> K = e+(e&8)+(L/16)*8
//   B 32x16 f16 : lane L -> col N=L%16, halves e=0..15 -> K = e+(e&8)+(L/16)*8
//   C/D 16x16 f32: lane L -> col N=L%16, vgpr r -> row M = r+(L/16)*8
// ---------------------------------------------------------------------------
__global__ __launch_bounds__(256) void gat_attn(
    const _Float16* __restrict__ hT, const float* __restrict__ f1g,
    const float* __restrict__ f2g, const float* __restrict__ f2maxg,
    const unsigned char* __restrict__ adj, float* __restrict__ out)
{
  __shared__ alignas(16) float    f2s[NN];           // 4 KB, pre-scaled by LOG2E
  __shared__ alignas(64) _Float16 Bs[2][FF * 32];    // 2 x 4 KB chunk [f][j_local]

  const int b    = blockIdx.x;
  const int t    = threadIdx.x;
  const int wave = t >> 5;
  const int lane = t & 31;
  const int half = lane >> 4;        // 0: lanes 0-15, 1: lanes 16-31
  const int lm   = lane & 15;
  const int i0   = blockIdx.y * 128 + wave * 16;
  const int i    = i0 + lm;          // row this lane owns (A / rowsum side)

  // stage f2[b][:]*LOG2E into LDS (256 threads x float4)
  {
    const v4f* src = (const v4f*)(f2g + b * NN);
    ((v4f*)f2s)[t] = src[t] * LOG2E;
  }

  const _Float16* hTb = hT + (size_t)b * FF * NN;
  const int sf = t >> 2, sj = (t & 3) * 8;   // this thread's staging slot
  // stage chunk 0 into buffer 0 (async; consumed after first barrier)
  async_b128(&hTb[(size_t)sf * NN + sj], &Bs[0][sf * 32 + sj]);

  const float fi = f1g[b * NN + i] * LOG2E;
  float c1, c2;   // fold the softmax shift through the leaky-relu max:
  {               // arg = max(f2+c1, fma(f2, SLOPE, c2)) = leakyrelu(f1+f2) - m
    float s  = fi + f2maxg[b] * LOG2E;
    float mi = fmaxf(s, SLOPE * s);          // upper bound on masked row max
    c1 = fi - mi;
    c2 = fmaf(SLOPE, fi, -mi);
  }

  const unsigned int* adj32 =
      (const unsigned int*)(adj + ((size_t)b * NN + i) * NN);

  v8f acc0 = {}, acc1 = {}, acc2 = {}, acc3 = {};
  float rowsum = 0.f;

  for (int c = 0; c < NN / 32; ++c) {
    const int j0 = c * 32;
    wait_async_lds();                // my async writes to Bs[c&1] are done
    __syncthreads();                 // everyone's are; prior reads of Bs[c^1] done

    if (c + 1 < NN / 32)             // prefetch next chunk into the other buffer
      async_b128(&hTb[(size_t)sf * NN + (j0 + 32) + sj],
                 &Bs[(c + 1) & 1][sf * 32 + sj]);

    if (j0 + 32 < NN)                // global_prefetch_b8 for adjacency stream
      __builtin_prefetch((const char*)adj32 + j0 + 32, 0, 1);

    // issue all loads up front so DS/VMEM latency hides under the exp block
    // adjacency bytes for the two 8-wide K runs, as dword pairs (0/1 bytes)
    const uint2 mA = *(const uint2*)(adj32 + (j0 >> 2) + 2 * half);       // j0+8h..
    const uint2 mB = *(const uint2*)(adj32 + (j0 >> 2) + 2 * half + 4);   // +16

    const v4f f2a0 = *(const v4f*)&f2s[j0 + half * 8];
    const v4f f2a1 = *(const v4f*)&f2s[j0 + half * 8 + 4];
    const v4f f2b0 = *(const v4f*)&f2s[j0 + half * 8 + 16];
    const v4f f2b1 = *(const v4f*)&f2s[j0 + half * 8 + 20];

    const _Float16* cur = Bs[c & 1];
    const v16h B0 = __builtin_shufflevector(*(const v8h*)&cur[(0 * 16 + lm) * 32 + half * 8], *(const v8h*)&cur[(0 * 16 + lm) * 32 + 16 + half * 8], 0, 1, 2, 3, 4, 5, 6, 7, 8, 9, 10, 11, 12, 13, 14, 15);
    const v16h B1 = __builtin_shufflevector(*(const v8h*)&cur[(1 * 16 + lm) * 32 + half * 8], *(const v8h*)&cur[(1 * 16 + lm) * 32 + 16 + half * 8], 0, 1, 2, 3, 4, 5, 6, 7, 8, 9, 10, 11, 12, 13, 14, 15);
    const v16h B2 = __builtin_shufflevector(*(const v8h*)&cur[(2 * 16 + lm) * 32 + half * 8], *(const v8h*)&cur[(2 * 16 + lm) * 32 + 16 + half * 8], 0, 1, 2, 3, 4, 5, 6, 7, 8, 9, 10, 11, 12, 13, 14, 15);
    const v16h B3 = __builtin_shufflevector(*(const v8h*)&cur[(3 * 16 + lm) * 32 + half * 8], *(const v8h*)&cur[(3 * 16 + lm) * 32 + 16 + half * 8], 0, 1, 2, 3, 4, 5, 6, 7, 8, 9, 10, 11, 12, 13, 14, 15);

    v16h A;
#pragma unroll
    for (int e = 0; e < 8; ++e) {       // K = e + half*8
      const float f2v = (e < 4 ? f2a0[e] : f2a1[e - 4]);
      const unsigned dw = (e < 4) ? mA.x : mA.y;
      const float bf = (float)((dw >> (8 * (e & 3))) & 0xffu);  // v_cvt_f32_ubyteN
      const float ex = __builtin_amdgcn_exp2f(
          fmaxf(f2v + c1, fmaf(f2v, SLOPE, c2)));
      const float p = ex * bf;
      A[e] = (_Float16)p;
      rowsum += p;
    }
#pragma unroll
    for (int e = 0; e < 8; ++e) {       // K = 16 + e + half*8
      const float f2v = (e < 4 ? f2b0[e] : f2b1[e - 4]);
      const unsigned dw = (e < 4) ? mB.x : mB.y;
      const float bf = (float)((dw >> (8 * (e & 3))) & 0xffu);
      const float ex = __builtin_amdgcn_exp2f(
          fmaxf(f2v + c1, fmaf(f2v, SLOPE, c2)));
      const float p = ex * bf;
      A[e + 8] = (_Float16)p;
      rowsum += p;
    }

    acc0 = __builtin_amdgcn_wmma_f32_16x16x32_f16(false, A, false, B0,
                                                  (short)0, acc0, false, false);
    acc1 = __builtin_amdgcn_wmma_f32_16x16x32_f16(false, A, false, B1,
                                                  (short)0, acc1, false, false);
    acc2 = __builtin_amdgcn_wmma_f32_16x16x32_f16(false, A, false, B2,
                                                  (short)0, acc2, false, false);
    acc3 = __builtin_amdgcn_wmma_f32_16x16x32_f16(false, A, false, B3,
                                                  (short)0, acc3, false, false);
  }

  // full row sum: lanes L and L+16 hold the two K-halves of row L%16
  const float S = rowsum + __shfl_xor(rowsum, 16, 32);

  // epilogue: divide by row sum, ELU, store (D vgpr r -> row r + half*8)
#pragma unroll
  for (int r = 0; r < 8; ++r) {
    const int m = r + half * 8;
    const float Sm  = __shfl(S, m, 32);      // lane m holds S for row m
    const float inv = 1.f / Sm;
    const size_t base = ((size_t)(b * NN + i0 + m)) * FF + lm;
    float v;
    v = acc0[r] * inv; out[base +  0] = v > 0.f ? v : __expf(v) - 1.f;
    v = acc1[r] * inv; out[base + 16] = v > 0.f ? v : __expf(v) - 1.f;
    v = acc2[r] * inv; out[base + 32] = v > 0.f ? v : __expf(v) - 1.f;
    v = acc3[r] * inv; out[base + 48] = v > 0.f ? v : __expf(v) - 1.f;
  }
}

// ---------------------------------------------------------------------------
// Workspace layout (needs ~4.5 MB):
//   [0, 4 MB)          hT   : f16, B x F x N (transposed h)
//   [4 MB, +128 KB)    f1   : f32, B*N
//   [+128 KB, +128 KB) f2   : f32, B*N
//   [+256 KB, +128 B)  f2max: f32, B
// ---------------------------------------------------------------------------
extern "C" void kernel_launch(void* const* d_in, const int* in_sizes, int n_in,
                              void* d_out, int out_size, void* d_ws, size_t ws_size,
                              hipStream_t stream) {
  const float*         X   = (const float*)d_in[0];          // (B,N,64) f32
  const unsigned char* adj = (const unsigned char*)d_in[1];  // (B,N,N) bool (1B)
  const float*         W   = (const float*)d_in[2];          // (64,64) f32
  const float*         a   = (const float*)d_in[3];          // (128,1) f32
  float* out = (float*)d_out;                                // (B,N,64) f32

  char* ws = (char*)d_ws;
  _Float16* hT    = (_Float16*)ws;
  float*    f1    = (float*)(ws + (size_t)4194304);
  float*    f2    = (float*)(ws + (size_t)4194304 + 131072);
  float*    f2max = (float*)(ws + (size_t)4194304 + 262144);

  gat_phase1<<<dim3((BB * NN) / 256), 256, 0, stream>>>(X, W, a, hT, f1, f2);
  gat_f2max <<<dim3(BB),             256, 0, stream>>>(f2, f2max);
  gat_attn  <<<dim3(BB, NN / 128),   256, 0, stream>>>(hT, f1, f2, f2max, adj, out);
}
